// Attention_2035814498461
// MI455X (gfx1250) — matrix operand order repaired and checked by reading, not compiled, not run on hardware
//
#include <hip/hip_runtime.h>
#include <cstddef>
#include <cstdint>

#define BATCH   2
#define SEQ     2048
#define DIM     1024
#define NHEADS  16
#define HD      64
#define SCALEF  0.125f   // 64^-0.5
#define KT      64       // GEMM K-tile

typedef float v8f  __attribute__((ext_vector_type(8)));
typedef __bf16 v16bf __attribute__((ext_vector_type(16)));
typedef unsigned short u16;
typedef u16 ushort8 __attribute__((ext_vector_type(8)));
typedef int v4i __attribute__((ext_vector_type(4)));

struct Pack32 { ushort8 lo, hi; };

static __device__ __forceinline__ v16bf mkfrag(ushort8 lo, ushort8 hi) {
  Pack32 p{lo, hi};
  return __builtin_bit_cast(v16bf, p);
}
static __device__ __forceinline__ u16 f2bf(float f) {
  unsigned u = __float_as_uint(f);
  return (u16)((u + 0x7FFFu + ((u >> 16) & 1u)) >> 16);   // RNE
}
static __device__ __forceinline__ v8f wmma_bf16(v16bf a, v16bf b, v8f c) {
  return __builtin_amdgcn_wmma_f32_16x16x32_bf16(false, a, false, b,
                                                 (short)0, c, false, false);
}

// ---- CDNA5 async global->LDS copy (ASYNCcnt-tracked), with safe fallback ----
typedef __attribute__((address_space(1))) v4i* gv4p;   // global int4*
typedef __attribute__((address_space(3))) v4i* lv4p;   // LDS int4*

static __device__ __forceinline__ void async_b128(const u16* g, u16* l) {
#if __has_builtin(__builtin_amdgcn_global_load_async_to_lds_b128)
  __builtin_amdgcn_global_load_async_to_lds_b128(
      (gv4p)(uintptr_t)g, (lv4p)(uintptr_t)l, 0, 0);
#else
  *(ushort8*)l = *(const ushort8*)g;   // sync fallback: global load + ds store
#endif
}
#if __has_builtin(__builtin_amdgcn_s_wait_asynccnt)
#define WAIT_ASYNC(n) __builtin_amdgcn_s_wait_asynccnt(n)
#else
#define WAIT_ASYNC(n) asm volatile("s_wait_asynccnt %0" :: "i"(n) : "memory")
#endif

// ---------------------------------------------------------------- convert
__global__ void __launch_bounds__(256)
cvt_f32_bf16(const float* __restrict__ src, u16* __restrict__ dst, int n) {
  int i = blockIdx.x * blockDim.x + threadIdx.x;
  if (i < n) dst[i] = f2bf(src[i]);
}

// ---------------------------------------------------------------- GEMM
// Y = X(4096x1024, bf16 rowmajor) @ W(1024x1024, bf16 rowmajor)^T
// Block tile 128x64, K-step 64, double-buffered LDS fed by async copies.
// mode 0: fp32 out + bias; mode 1: bf16 (b,h,n,d); mode 2: bf16 (b,h,d,n)
__global__ void __launch_bounds__(256)
gemm_bf16(const u16* __restrict__ X, const u16* __restrict__ W,
          float* __restrict__ outF, u16* __restrict__ outB,
          const float* __restrict__ bias, int mode) {
  __shared__ u16 As[2][128 * KT];   // 2 x 16 KB
  __shared__ u16 Bs[2][64 * KT];    // 2 x  8 KB
  const int tid  = threadIdx.x;
  const int wave = tid >> 5;
  const int lane = tid & 31;
  const int hs   = lane >> 4;
  const int l15  = lane & 15;
  const int mblk = blockIdx.x & 31;       // 4096/128
  const int nblk = blockIdx.x >> 5;       // 1024/64
  const int row0 = mblk * 128;
  const int col0 = nblk * 64;

  auto issue_tile = [&](int buf, int k0) {
#pragma unroll
    for (int i = 0; i < 4; ++i) {         // A: 1024 chunks, 4/thread
      const int c = tid + i * 256;
      const int r = c >> 3, kc = (c & 7) * 8;
      async_b128(X + (size_t)(row0 + r) * DIM + k0 + kc, &As[buf][r * KT + kc]);
    }
#pragma unroll
    for (int i = 0; i < 2; ++i) {         // B: 512 chunks, 2/thread
      const int c = tid + i * 256;
      const int col = c >> 3, kc = (c & 7) * 8;
      async_b128(W + (size_t)(col0 + col) * DIM + k0 + kc, &Bs[buf][col * KT + kc]);
    }
  };

  const v8f vzero = {0.f,0.f,0.f,0.f,0.f,0.f,0.f,0.f};
  v8f acc[4] = {vzero, vzero, vzero, vzero};

  issue_tile(0, 0);
  const int NKT = DIM / KT;               // 16
  for (int kt = 0; kt < NKT; ++kt) {
    const int cur = kt & 1;
    if (kt + 1 < NKT) {
      issue_tile(cur ^ 1, (kt + 1) * KT);
      WAIT_ASYNC(6);
    } else {
      WAIT_ASYNC(0);
    }
    __syncthreads();

    const u16* Arow = &As[cur][(wave * 16 + l15) * KT];
#pragma unroll
    for (int kk = 0; kk < KT; kk += 32) {
      const int ka = kk + (hs ? 8 : 0);
      v16bf a = mkfrag(*(const ushort8*)(Arow + ka),
                       *(const ushort8*)(Arow + ka + 16));
      const int kb = kk + (hs ? 8 : 0);
#pragma unroll
      for (int t = 0; t < 4; ++t) {
        const u16* Brow = &Bs[cur][(t * 16 + l15) * KT + kb];
        v16bf b = mkfrag(*(const ushort8*)(Brow),
                         *(const ushort8*)(Brow + 16));
        acc[t] = wmma_bf16(a, b, acc[t]);
      }
    }
    __syncthreads();
  }

  const int wrow0 = row0 + wave * 16;
#pragma unroll
  for (int t = 0; t < 4; ++t) {
    const int col = col0 + t * 16 + l15;
#pragma unroll
    for (int r = 0; r < 8; ++r) {
      const int row = wrow0 + r + (hs ? 8 : 0);
      const float v = acc[t][r];
      if (mode == 0) {
        outF[(size_t)row * DIM + col] = v + bias[col];
      } else {
        const int b = row >> 11, n = row & 2047;
        const int h = col >> 6,  d = col & 63;
        size_t idx = (mode == 1)
            ? ((size_t)((b * NHEADS + h) * SEQ + n) * HD + d)
            : ((size_t)((b * NHEADS + h) * HD + d) * SEQ + n);
        outB[idx] = f2bf(v);
      }
    }
  }
}

// ---------------------------------------------------------------- attention
// Block = 4 waves = 64 query rows of one (b,h); one wave owns 16 rows.
// K and V^T tiles staged in LDS via async copies, double-buffered.
__global__ void __launch_bounds__(128)
attn_kernel(const u16* __restrict__ Qh, const u16* __restrict__ Kh,
            const u16* __restrict__ Vt, float* __restrict__ attn,
            u16* __restrict__ Obf, float* __restrict__ invb,
            const int* __restrict__ maskp) {
  __shared__ u16 Ks[2][64 * HD];    // 2 x 8 KB (64 keys x 64 d)
  __shared__ u16 Vs[2][64 * 64];    // 2 x 8 KB (64 d x 64 m)
  __shared__ float sinv[4][16];
  const int use_mask = *maskp;
  const int tid  = threadIdx.x;
  const int wave = tid >> 5;
  const int lane = tid & 31;
  const int hs   = lane >> 4;
  const int l15  = lane & 15;
  const int bh   = blockIdx.x >> 5;     // SEQ/64 = 32 row-blocks per head
  const int nblk = blockIdx.x & 31;
  const int n0   = nblk * 64 + wave * 16;

  const u16* q  = Qh + (size_t)bh * SEQ * HD;
  const u16* k  = Kh + (size_t)bh * SEQ * HD;
  const u16* vt = Vt + (size_t)bh * HD * SEQ;
  float* arow   = attn + (size_t)bh * SEQ * SEQ;

  auto issueK = [&](int buf, int m0) {  // 512 chunks, 4/thread
#pragma unroll
    for (int i = 0; i < 4; ++i) {
      const int c = tid + i * 128;
      const int r = c >> 3, kc = (c & 7) * 8;
      async_b128(k + (size_t)(m0 + r) * HD + kc, &Ks[buf][r * HD + kc]);
    }
  };
  auto issueV = [&](int buf, int m0) {  // 512 chunks, 4/thread
#pragma unroll
    for (int i = 0; i < 4; ++i) {
      const int c = tid + i * 128;
      const int r = c >> 3, mc = (c & 7) * 8;
      async_b128(vt + (size_t)r * SEQ + m0 + mc, &Vs[buf][r * 64 + mc]);
    }
  };

  // ---- Pass A: logits S = (Q K^T) * scale, causal mask ----
  const u16* qrow = q + (size_t)(n0 + l15) * HD + (hs ? 8 : 0);
  const v16bf aq0 = mkfrag(*(const ushort8*)(qrow),
                           *(const ushort8*)(qrow + 16));
  const v16bf aq1 = mkfrag(*(const ushort8*)(qrow + 32),
                           *(const ushort8*)(qrow + 48));
  const v8f vzero = {0.f,0.f,0.f,0.f,0.f,0.f,0.f,0.f};

  issueK(0, 0);
  for (int it = 0; it < SEQ / 64; ++it) {
    const int cur = it & 1;
    if (it + 1 < SEQ / 64) { issueK(cur ^ 1, (it + 1) * 64); WAIT_ASYNC(4); }
    else                   { WAIT_ASYNC(0); }
    __syncthreads();
#pragma unroll
    for (int sub = 0; sub < 4; ++sub) {
      const int m0 = it * 64 + sub * 16;
      const u16* krow = &Ks[cur][(sub * 16 + l15) * HD + (hs ? 8 : 0)];
      v16bf b0 = mkfrag(*(const ushort8*)(krow),      *(const ushort8*)(krow + 16));
      v16bf b1 = mkfrag(*(const ushort8*)(krow + 32), *(const ushort8*)(krow + 48));
      v8f s = vzero;
      s = wmma_bf16(aq0, b0, s);
      s = wmma_bf16(aq1, b1, s);
      const int col = m0 + l15;
#pragma unroll
      for (int r = 0; r < 8; ++r) {
        const int row = n0 + r + (hs ? 8 : 0);
        float v = s[r] * SCALEF;
        if (use_mask && col > row) v = -1.0e9f;
        arow[(size_t)row * SEQ + col] = v;
      }
    }
    __syncthreads();
  }

  // ---- Pass B: row max (lane pair (l, l+16) splits the row) ----
  float* prw = arow + (size_t)(n0 + l15) * SEQ;
  float rmx = -3.0e38f;
  {
    const float* pr = prw + (hs ? SEQ / 2 : 0);
    for (int i = 0; i < SEQ / 2; i += 4) {
      float4 v = *(const float4*)(pr + i);
      rmx = fmaxf(rmx, fmaxf(fmaxf(v.x, v.y), fmaxf(v.z, v.w)));
    }
    rmx = fmaxf(rmx, __shfl_xor(rmx, 16, 32));
  }

  // ---- Pass C: exp + row sum + write P + O = P V (bf16 WMMA) ----
  v8f o[4] = {vzero, vzero, vzero, vzero};
  float lsum = 0.f;
  issueV(0, 0);
  for (int it = 0; it < SEQ / 64; ++it) {
    const int cur = it & 1;
    if (it + 1 < SEQ / 64) { issueV(cur ^ 1, (it + 1) * 64); WAIT_ASYNC(4); }
    else                   { WAIT_ASYNC(0); }
    __syncthreads();
#pragma unroll
    for (int sub = 0; sub < 2; ++sub) {
      const int m0 = it * 64 + sub * 32;
      const int c0 = m0 + (hs ? 8 : 0);
      const int c1 = c0 + 16;
      float4 pa = *(const float4*)(prw + c0);
      float4 pb = *(const float4*)(prw + c0 + 4);
      float4 pc = *(const float4*)(prw + c1);
      float4 pd = *(const float4*)(prw + c1 + 4);
      float e0 = __expf(pa.x - rmx), e1 = __expf(pa.y - rmx);
      float e2 = __expf(pa.z - rmx), e3 = __expf(pa.w - rmx);
      float e4 = __expf(pb.x - rmx), e5 = __expf(pb.y - rmx);
      float e6 = __expf(pb.z - rmx), e7 = __expf(pb.w - rmx);
      float e8 = __expf(pc.x - rmx), e9 = __expf(pc.y - rmx);
      float ea = __expf(pc.z - rmx), eb = __expf(pc.w - rmx);
      float ec = __expf(pd.x - rmx), ed = __expf(pd.y - rmx);
      float ee = __expf(pd.z - rmx), ef = __expf(pd.w - rmx);
      lsum += (e0+e1+e2+e3)+(e4+e5+e6+e7)+(e8+e9+ea+eb)+(ec+ed+ee+ef);
      *(float4*)(prw + c0)     = make_float4(e0, e1, e2, e3);
      *(float4*)(prw + c0 + 4) = make_float4(e4, e5, e6, e7);
      *(float4*)(prw + c1)     = make_float4(e8, e9, ea, eb);
      *(float4*)(prw + c1 + 4) = make_float4(ec, ed, ee, ef);
      ushort8 lo = {f2bf(e0), f2bf(e1), f2bf(e2), f2bf(e3),
                    f2bf(e4), f2bf(e5), f2bf(e6), f2bf(e7)};
      ushort8 hi = {f2bf(e8), f2bf(e9), f2bf(ea), f2bf(eb),
                    f2bf(ec), f2bf(ed), f2bf(ee), f2bf(ef)};
      v16bf pf = mkfrag(lo, hi);
#pragma unroll
      for (int t = 0; t < 4; ++t) {
        const u16* vr = &Vs[cur][(t * 16 + l15) * 64 + sub * 32 + (hs ? 8 : 0)];
        v16bf vb = mkfrag(*(const ushort8*)(vr), *(const ushort8*)(vr + 16));
        o[t] = wmma_bf16(pf, vb, o[t]);
      }
    }
    __syncthreads();
  }

  lsum += __shfl_xor(lsum, 16, 32);
  const float inv = 1.0f / lsum;
  if (lane < 16) {
    invb[(size_t)bh * SEQ + n0 + lane] = inv;
    sinv[wave][lane] = inv;
  }
  __syncthreads();

  const int b = bh >> 4, h = bh & 15;
#pragma unroll
  for (int t = 0; t < 4; ++t) {
    const int col = h * HD + t * 16 + l15;
#pragma unroll
    for (int r = 0; r < 8; ++r) {
      const int row = n0 + r + (hs ? 8 : 0);
      const float iv = sinv[wave][r + (hs ? 8 : 0)];
      Obf[(size_t)(b * SEQ + row) * DIM + col] = f2bf(o[t][r] * iv);
    }
  }
}

// ---------------------------------------------------------------- normalize
__global__ void __launch_bounds__(256)
normalize_attn(float* __restrict__ attn, const float* __restrict__ invb) {
  size_t i = ((size_t)blockIdx.x * blockDim.x + threadIdx.x) * 4;
  float4 v = *(float4*)(attn + i);
  const float s = invb[i >> 11];   // row = i / SEQ
  v.x *= s; v.y *= s; v.z *= s; v.w *= s;
  *(float4*)(attn + i) = v;
}

// ---------------------------------------------------------------- launch
extern "C" void kernel_launch(void* const* d_in, const int* in_sizes, int n_in,
                              void* d_out, int out_size, void* d_ws, size_t ws_size,
                              hipStream_t stream) {
  const float* q      = (const float*)d_in[0];
  const float* k      = (const float*)d_in[1];
  const float* v      = (const float*)d_in[2];
  const float* qkv_w  = (const float*)d_in[3];
  const float* proj_w = (const float*)d_in[4];
  const float* proj_b = (const float*)d_in[5];
  const int*   msk    = (const int*)d_in[6];

  float* x_out    = (float*)d_out;
  float* attn_out = x_out + (size_t)BATCH * SEQ * DIM;

  const size_t MC = (size_t)BATCH * SEQ * DIM;      // 4,194,304
  u16* qbf  = (u16*)d_ws;
  u16* kbf  = qbf + MC;
  u16* vbf  = kbf + MC;
  u16* wbf  = vbf + MC;                              // 3*C*C
  u16* pwbf = wbf + (size_t)3 * DIM * DIM;
  u16* qh   = pwbf + (size_t)DIM * DIM;
  u16* kh   = qh + MC;
  u16* vt   = kh + MC;
  u16* obf  = vt + MC;
  float* invb = (float*)(obf + MC);                  // B*H*SEQ floats

  // 1) fp32 -> bf16
  cvt_f32_bf16<<<(int)(MC / 256), 256, 0, stream>>>(q, qbf, (int)MC);
  cvt_f32_bf16<<<(int)(MC / 256), 256, 0, stream>>>(k, kbf, (int)MC);
  cvt_f32_bf16<<<(int)(MC / 256), 256, 0, stream>>>(v, vbf, (int)MC);
  cvt_f32_bf16<<<(3 * DIM * DIM) / 256, 256, 0, stream>>>(qkv_w, wbf, 3 * DIM * DIM);
  cvt_f32_bf16<<<(DIM * DIM) / 256, 256, 0, stream>>>(proj_w, pwbf, DIM * DIM);

  // 2) QKV projections (WMMA bf16, async-LDS staged)
  gemm_bf16<<<512, 256, 0, stream>>>(qbf, wbf,                         nullptr, qh, nullptr, 1);
  gemm_bf16<<<512, 256, 0, stream>>>(kbf, wbf + (size_t)DIM * DIM,     nullptr, kh, nullptr, 1);
  gemm_bf16<<<512, 256, 0, stream>>>(vbf, wbf + (size_t)2 * DIM * DIM, nullptr, vt, nullptr, 2);

  // 3) attention core (S, softmax stats, unnormalized P, O = P V)
  attn_kernel<<<BATCH * NHEADS * (SEQ / 64), 128, 0, stream>>>(
      qh, kh, vt, attn_out, obf, invb, msk);

  // 4) normalize attn probabilities
  normalize_attn<<<(int)(((size_t)BATCH * NHEADS * SEQ * SEQ) / (4 * 256)), 256, 0, stream>>>(
      attn_out, invb);

  // 5) output projection + bias
  gemm_bf16<<<512, 256, 0, stream>>>(obf, pwbf, x_out, nullptr, proj_b, 0);
}
